// StackedLSTMClassifier_48911087567700
// MI455X (gfx1250) — compile-verified, matrix operand order checked by reading, not run on hardware
//
#include <hip/hip_runtime.h>
#include <hip/hip_bf16.h>

#define BATCH 8192
#define TSTEPS 20
#define IN 16
#define H1V 128
#define H2V 64
#define FCV 32

typedef __attribute__((ext_vector_type(16))) _Float16 v16h;
typedef __attribute__((ext_vector_type(8)))  float    v8f;

// ---- workspace byte offsets (pre-swizzled f16 weight fragments) ----
// fragment = 512 halfs = 1024 B, layout: half index = lane*16 + e
//   n_local = lane & 15 ; k_local = (e&7) + 8*((e>>3)*2 + (lane>>4))
#define WS_WIH1 0           //  32 frags (ct=0..31, K padded 16->32)
#define WS_WHH1 32768       // 128 frags (ct*4 + kc)
#define WS_WIH2 163840      //  64 frags (ct*4 + kc)
#define WS_WHH2 229376      //  32 frags (ct*2 + kc)
#define WS_B1   262144      // 512 f32 (bih1+bhh1)
#define WS_B2   264192      // 256 f32 (bih2+bhh2)
#define WS_END  265216

#define LDS_STAGE 265216            // + grp*6144 : stage1(16x128 f16) + stage2(16x64 f16)
#define LDS_TOTAL (265216 + 2*6144) // 277504 B  (<= 320KB/WGP on gfx1250)

// ---------------- prep: f32 weights -> f16 WMMA B-fragments in d_ws ----------------
__global__ void prep_weights(const float* __restrict__ Wih1, const float* __restrict__ Whh1,
                             const float* __restrict__ bih1, const float* __restrict__ bhh1,
                             const float* __restrict__ Wih2, const float* __restrict__ Whh2,
                             const float* __restrict__ bih2, const float* __restrict__ bhh2,
                             char* __restrict__ ws)
{
    int idx = blockIdx.x * blockDim.x + threadIdx.x;
    if (idx < 131072) {
        int within = idx & 511;
        int L  = within >> 4;
        int e  = within & 15;
        int nl = L & 15;
        int kl = (e & 7) + 8 * (((e >> 3) << 1) + (L >> 4));   // 0..31
        float val;
        if (idx < 16384) {                       // WIH1: 512x16, K padded to 32
            int ct = idx >> 9;
            int n  = 16 * ct + nl;
            val = (kl < 16) ? Wih1[n * 16 + kl] : 0.f;
        } else if (idx < 81920) {                // WHH1: 512x128
            int f  = (idx - 16384) >> 9;
            int ct = f >> 2, kc = f & 3;
            int n  = 16 * ct + nl, k = 32 * kc + kl;
            val = Whh1[n * 128 + k];
        } else if (idx < 114688) {               // WIH2: 256x128
            int f  = (idx - 81920) >> 9;
            int ct = f >> 2, kc = f & 3;
            int n  = 16 * ct + nl, k = 32 * kc + kl;
            val = Wih2[n * 128 + k];
        } else {                                 // WHH2: 256x64
            int f  = (idx - 114688) >> 9;
            int ct = f >> 1, kc = f & 1;
            int n  = 16 * ct + nl, k = 32 * kc + kl;
            val = Whh2[n * 64 + k];
        }
        ((_Float16*)ws)[idx] = (_Float16)val;
    } else if (idx < 131072 + 512) {
        int n = idx - 131072;
        ((float*)(ws + WS_B1))[n] = bih1[n] + bhh1[n];
    } else if (idx < 131072 + 768) {
        int n = idx - 131072 - 512;
        ((float*)(ws + WS_B2))[n] = bih2[n] + bhh2[n];
    }
}

// ---------------- helpers ----------------
static __device__ __forceinline__ v8f wmma16(v16h a, v16h b, v8f c) {
    return __builtin_amdgcn_wmma_f32_16x16x32_f16(false, a, false, b, (short)0, c, false, false);
}
static __device__ __forceinline__ v16h ld_bfrag(const _Float16* fragBase, int lane) {
    return *(const v16h*)(fragBase + lane * 16);     // 2x ds_load_b128
}
// A-fragment from row-major f16 staging: p = row base (already at M*stride + kBase + 8*(lane>>4))
static __device__ __forceinline__ v16h ld_afrag(const _Float16* p) {
    v16h r;
#pragma unroll
    for (int e = 0; e < 8; ++e) { r[e] = p[e]; r[8 + e] = p[16 + e]; }
    return r;
}
static __device__ __forceinline__ float sigm(float v) { return 1.f / (1.f + __expf(-v)); }
static __device__ __forceinline__ v8f zero8() {
    v8f z;
#pragma unroll
    for (int r = 0; r < 8; ++r) z[r] = 0.f;
    return z;
}

// ---------------- fused 2-layer LSTM + LN + MLP head ----------------
__global__ __launch_bounds__(256, 1)
void lstm_fused(const float* __restrict__ x, const char* __restrict__ ws,
                const float* __restrict__ gamma, const float* __restrict__ beta,
                const float* __restrict__ W1, const float* __restrict__ b1,
                const float* __restrict__ W2, const float* __restrict__ b2,
                float* __restrict__ out)
{
    extern __shared__ __align__(16) char smem[];
    const int tid = threadIdx.x;

    // cooperative load of all swizzled weights + biases into LDS (259 KB)
    {
        const uint4* src = (const uint4*)ws;
        uint4* dst = (uint4*)smem;
        for (int i = tid; i < WS_END / 16; i += blockDim.x) dst[i] = src[i];
    }
    __syncthreads();

    const _Float16* sWIH1 = (const _Float16*)(smem + WS_WIH1);
    const _Float16* sWHH1 = (const _Float16*)(smem + WS_WHH1);
    const _Float16* sWIH2 = (const _Float16*)(smem + WS_WIH2);
    const _Float16* sWHH2 = (const _Float16*)(smem + WS_WHH2);
    const float* sB1 = (const float*)(smem + WS_B1);
    const float* sB2 = (const float*)(smem + WS_B2);

    const int lane = tid & 31;
    const int wave = tid >> 5;
    const int grp  = wave >> 2;      // 2 groups of 4 waves; each group owns 16 batch rows
    const int wl   = wave & 3;       // wave-in-group: splits gate-column tiles
    _Float16* stage1 = (_Float16*)(smem + LDS_STAGE + grp * 6144);   // 16 x 128 f16
    _Float16* stage2 = stage1 + 16 * H1V;                            // 16 x 64  f16

    const int r0   = blockIdx.x * 32 + grp * 16;
    const int mrow = lane & 15;       // A fragment row (M); C fragment column (N)
    const int half = lane >> 4;       // lane half selects K/M sub-block
    const int mbase = half * 8;       // C-layout row base: m = r + 8*(lane>>4)

    // recurrent state
    v16h a1[4], a2[2];                // h1 (16x128), h2 (16x64) as f16 A-fragments
    v8f  c1t[2], c2t;                 // c1 tiles j=2*wl,2*wl+1 ; c2 tile j2=wl (f32 C-layout)
#pragma unroll
    for (int i = 0; i < 4; ++i)
#pragma unroll
        for (int e = 0; e < 16; ++e) a1[i][e] = (_Float16)0.f;
#pragma unroll
    for (int i = 0; i < 2; ++i)
#pragma unroll
        for (int e = 0; e < 16; ++e) a2[i][e] = (_Float16)0.f;
    c1t[0] = zero8(); c1t[1] = zero8(); c2t = zero8();

    for (int t = 0; t < TSTEPS; ++t) {
        // ---- x_t fragment (K=16 real, padded to 32) ----
        const float* xrow = x + (r0 + mrow) * (TSTEPS * IN) + t * IN + half * 8;
        v16h ax;
#pragma unroll
        for (int e = 0; e < 8; ++e) { ax[e] = (_Float16)xrow[e]; ax[8 + e] = (_Float16)0.f; }

        // ======== LSTM layer 1: this wave computes hidden tiles j = 2*wl, 2*wl+1 ========
#pragma unroll
        for (int jj = 0; jj < 2; ++jj) {
            const int j = 2 * wl + jj;
            v8f acc[4];
#pragma unroll
            for (int g = 0; g < 4; ++g) {
                const int ct = g * 8 + j;                         // z-column tile (512 cols)
                v16h bx  = ld_bfrag(sWIH1 + ct * 512, lane);
                v16h bh0 = ld_bfrag(sWHH1 + (ct * 4 + 0) * 512, lane);
                v16h bh1 = ld_bfrag(sWHH1 + (ct * 4 + 1) * 512, lane);
                v16h bh2 = ld_bfrag(sWHH1 + (ct * 4 + 2) * 512, lane);
                v16h bh3 = ld_bfrag(sWHH1 + (ct * 4 + 3) * 512, lane);
                v8f c = wmma16(ax, bx, zero8());                  // C starts as inline 0
                c = wmma16(a1[0], bh0, c);
                c = wmma16(a1[1], bh1, c);
                c = wmma16(a1[2], bh2, c);
                c = wmma16(a1[3], bh3, c);
                acc[g] = c;
            }
            const int col = 16 * j + (lane & 15);
            const float bvi = sB1[0 * 128 + col];
            const float bvf = sB1[1 * 128 + col];
            const float bvg = sB1[2 * 128 + col];
            const float bvo = sB1[3 * 128 + col];
#pragma unroll
            for (int r = 0; r < 8; ++r) {
                float ig = sigm(acc[0][r] + bvi);
                float fg = sigm(acc[1][r] + bvf);
                float gg = tanhf(acc[2][r] + bvg);
                float og = sigm(acc[3][r] + bvo);
                float cv = fg * c1t[jj][r] + ig * gg;
                c1t[jj][r] = cv;
                stage1[(mbase + r) * H1V + col] = (_Float16)(og * tanhf(cv));
            }
        }
        __syncthreads();
        // all waves reload full h1 as A-fragments
#pragma unroll
        for (int kc = 0; kc < 4; ++kc)
            a1[kc] = ld_afrag(stage1 + mrow * H1V + 32 * kc + half * 8);

        // ======== LSTM layer 2: this wave computes hidden tile j2 = wl ========
        {
            const int j2 = wl;
            v8f acc[4];
#pragma unroll
            for (int g = 0; g < 4; ++g) {
                const int ct = g * 4 + j2;                        // z-column tile (256 cols)
                v16h bi0 = ld_bfrag(sWIH2 + (ct * 4 + 0) * 512, lane);
                v16h bi1 = ld_bfrag(sWIH2 + (ct * 4 + 1) * 512, lane);
                v16h bi2 = ld_bfrag(sWIH2 + (ct * 4 + 2) * 512, lane);
                v16h bi3 = ld_bfrag(sWIH2 + (ct * 4 + 3) * 512, lane);
                v16h bh0 = ld_bfrag(sWHH2 + (ct * 2 + 0) * 512, lane);
                v16h bh1 = ld_bfrag(sWHH2 + (ct * 2 + 1) * 512, lane);
                v8f c = wmma16(a1[0], bi0, zero8());              // C starts as inline 0
                c = wmma16(a1[1], bi1, c);
                c = wmma16(a1[2], bi2, c);
                c = wmma16(a1[3], bi3, c);
                c = wmma16(a2[0], bh0, c);
                c = wmma16(a2[1], bh1, c);
                acc[g] = c;
            }
            const int col = 16 * j2 + (lane & 15);
            const float bvi = sB2[0 * 64 + col];
            const float bvf = sB2[1 * 64 + col];
            const float bvg = sB2[2 * 64 + col];
            const float bvo = sB2[3 * 64 + col];
#pragma unroll
            for (int r = 0; r < 8; ++r) {
                float ig = sigm(acc[0][r] + bvi);
                float fg = sigm(acc[1][r] + bvf);
                float gg = tanhf(acc[2][r] + bvg);
                float og = sigm(acc[3][r] + bvo);
                float cv = fg * c2t[r] + ig * gg;
                c2t[r] = cv;
                stage2[(mbase + r) * H2V + col] = (_Float16)(og * tanhf(cv));
            }
        }
        __syncthreads();
#pragma unroll
        for (int kc = 0; kc < 2; ++kc)
            a2[kc] = ld_afrag(stage2 + mrow * H2V + 32 * kc + half * 8);
    }

    // ======== head: LayerNorm(H2) -> FC(32)+ReLU -> FC(1), one lane per batch row ========
    if (tid < 32) {
        const int g = tid >> 4, m = tid & 15;
        const _Float16* st2 = (const _Float16*)(smem + LDS_STAGE + g * 6144) + 16 * H1V;
        float v[H2V];
        float mean = 0.f;
#pragma unroll
        for (int k = 0; k < H2V; ++k) { v[k] = (float)st2[m * H2V + k]; mean += v[k]; }
        mean *= (1.f / H2V);
        float var = 0.f;
#pragma unroll
        for (int k = 0; k < H2V; ++k) { float d = v[k] - mean; var += d * d; }
        var *= (1.f / H2V);
        const float rstd = rsqrtf(var + 1e-5f);
#pragma unroll
        for (int k = 0; k < H2V; ++k) v[k] = (v[k] - mean) * rstd * gamma[k] + beta[k];
        float acc2 = b2[0];
        for (int o = 0; o < FCV; ++o) {
            float s = b1[o];
#pragma unroll
            for (int k = 0; k < H2V; ++k) s += v[k] * W1[o * H2V + k];
            acc2 += fmaxf(s, 0.f) * W2[o];
        }
        out[blockIdx.x * 32 + g * 16 + m] = acc2;
    }
}

extern "C" void kernel_launch(void* const* d_in, const int* in_sizes, int n_in,
                              void* d_out, int out_size, void* d_ws, size_t ws_size,
                              hipStream_t stream)
{
    const float* x     = (const float*)d_in[0];
    const float* Wih1  = (const float*)d_in[1];
    const float* Whh1  = (const float*)d_in[2];
    const float* bih1  = (const float*)d_in[3];
    const float* bhh1  = (const float*)d_in[4];
    const float* Wih2  = (const float*)d_in[5];
    const float* Whh2  = (const float*)d_in[6];
    const float* bih2  = (const float*)d_in[7];
    const float* bhh2  = (const float*)d_in[8];
    const float* gamma = (const float*)d_in[9];
    const float* beta  = (const float*)d_in[10];
    const float* W1    = (const float*)d_in[11];
    const float* b1    = (const float*)d_in[12];
    const float* W2    = (const float*)d_in[13];
    const float* b2    = (const float*)d_in[14];
    float* out = (float*)d_out;
    char* ws = (char*)d_ws;

    const int prepThreads = 131072 + 768;
    prep_weights<<<(prepThreads + 255) / 256, 256, 0, stream>>>(
        Wih1, Whh1, bih1, bhh1, Wih2, Whh2, bih2, bhh2, ws);

    lstm_fused<<<BATCH / 32, 256, LDS_TOTAL, stream>>>(
        x, ws, gamma, beta, W1, b1, W2, b2, out);
}
